// pre_embedding_81535659148048
// MI455X (gfx1250) — matrix operand order repaired and checked by reading, not compiled, not run on hardware
//
#include <hip/hip_runtime.h>

#define N_NODES 50000
#define N_EDGES 800000
#define DIM 128
#define N_GRAPHS 64
#define N_CLASS 10
#define HID 64

static_assert(N_NODES % 16 == 0, "GEMM kernel assumes whole 16-row tiles");

typedef __attribute__((ext_vector_type(16))) _Float16 v16h;
typedef __attribute__((ext_vector_type(8)))  float    v8f;

// ---------------------------------------------------------------- utilities
// n4 = number of float4 elements (all buffers we zero are divisible by 4).
__global__ __launch_bounds__(256) void fill_zero4_kernel(float4* __restrict__ p, int n4) {
  int i = blockIdx.x * 256 + threadIdx.x;
  if (i < n4) p[i] = make_float4(0.f, 0.f, 0.f, 0.f);
}

__global__ __launch_bounds__(256) void degree_kernel(const int* __restrict__ src,
                                                     const int* __restrict__ dst,
                                                     float* __restrict__ deg_out,
                                                     float* __restrict__ deg_in,
                                                     int n_edges) {
  int e = blockIdx.x * 256 + threadIdx.x;
  if (e >= n_edges) return;
  atomicAdd(&deg_out[src[e]], 1.0f);
  atomicAdd(&deg_in[dst[e]], 1.0f);
}

__global__ __launch_bounds__(256) void norm_kernel(float* __restrict__ a,
                                                   float* __restrict__ b, int n) {
  int i = blockIdx.x * 256 + threadIdx.x;
  if (i >= n) return;
  a[i] = rsqrtf(fmaxf(a[i], 1.0f));
  b[i] = rsqrtf(fmaxf(b[i], 1.0f));
}

// ------------------------------------------------- edge gather + scatter-add
// One wave per edge; each lane moves 4 contiguous floats (32 lanes * 4 = 128).
// ns == nullptr -> unscaled copy_u/sum aggregation (final h_agg pass).
__global__ __launch_bounds__(256) void edge_scatter_kernel(const int* __restrict__ src,
                                                           const int* __restrict__ dst,
                                                           const float* __restrict__ ns,
                                                           const float* __restrict__ h,
                                                           float* __restrict__ agg,
                                                           int n_edges) {
  int t = blockIdx.x * 256 + threadIdx.x;
  int e = t >> 5;
  if (e >= n_edges) return;
  int q = (t & 31) * 4;
  int s = src[e];
  int d = dst[e];
  float scale = (ns != nullptr) ? ns[s] : 1.0f;
  const float4 x = *reinterpret_cast<const float4*>(h + (size_t)s * DIM + q);
  float* out = agg + (size_t)d * DIM + q;
  atomicAdd(out + 0, x.x * scale);
  atomicAdd(out + 1, x.y * scale);
  atomicAdd(out + 2, x.z * scale);
  atomicAdd(out + 3, x.w * scale);
}

// ----------------------- fused (agg * nd) @ W + b, ReLU  -- WMMA f16->f32
// Block = 256 threads (8 waves). Block computes 16 rows x 128 cols, in place.
// Wave w computes the 16x16 tile at columns [16w, 16w+16).
// Grid is exactly N/16 blocks (N % 16 == 0): no bounds checks anywhere.
__global__ __launch_bounds__(256) void gemm_norm_bias_relu_kernel(
    const float* __restrict__ W,     // [DIM, DIM] row-major (k, j)
    const float* __restrict__ bias,  // [DIM]
    const float* __restrict__ nd,    // [N]
    float* __restrict__ agg) {       // [N, DIM] read (X) then overwritten (out)
  __shared__ _Float16 As[16][DIM];    //  4 KB : 16 rows of scaled input, f16
  __shared__ _Float16 WsT[DIM][DIM];  // 32 KB : W transposed -> WsT[j][k]
  __shared__ float    Outs[16][DIM];  //  8 KB : D tile for coalesced store
  const int row0 = blockIdx.x * 16;
  const int tid  = threadIdx.x;

  // Stage A (scaled by nd, cast f16): float4 loads, ds_store_b64 per quad.
  for (int i = tid; i < 16 * (DIM / 4); i += 256) {
    int r = i >> 5, c4 = (i & 31) * 4;
    int gr = row0 + r;
    const float4 x = *reinterpret_cast<const float4*>(agg + (size_t)gr * DIM + c4);
    float s = nd[gr];
    As[r][c4 + 0] = (_Float16)(x.x * s);
    As[r][c4 + 1] = (_Float16)(x.y * s);
    As[r][c4 + 2] = (_Float16)(x.z * s);
    As[r][c4 + 3] = (_Float16)(x.w * s);
  }
  // Stage W transposed (so B-fragment reads are contiguous along K).
  for (int i = tid; i < DIM * (DIM / 4); i += 256) {
    int k = i >> 5, j4 = (i & 31) * 4;
    const float4 w = *reinterpret_cast<const float4*>(W + (size_t)k * DIM + j4);
    WsT[j4 + 0][k] = (_Float16)w.x;
    WsT[j4 + 1][k] = (_Float16)w.y;
    WsT[j4 + 2][k] = (_Float16)w.z;
    WsT[j4 + 3][k] = (_Float16)w.w;
  }
  __syncthreads();

  const int wave = tid >> 5;        // N tile index 0..7
  const int lane = tid & 31;
  const int hi   = lane >> 4;       // lane half
  const int lm   = lane & 15;       // A-frag: M ; B/C/D-frag: N
  const int ncol = wave * 16 + lm;

  // C/D tile: lane holds column ncol, rows M = 8*hi + v in element v.
  v8f acc;
  const float bv = bias[ncol];
#pragma unroll
  for (int v = 0; v < 8; ++v) acc[v] = bv;

#pragma unroll
  for (int k0 = 0; k0 < DIM; k0 += 32) {
    v16h afrag, bfrag;
    // A 16x32 f16, this code's contiguous k order: lanes 0-15 hold K = 0..15, lanes 16-31 hold K = 16..31
#pragma unroll
    for (int i = 0; i < 8; ++i) {
      afrag[i]     = As[lm][k0 + 16 * hi + i];
      afrag[i + 8] = As[lm][k0 + 16 * hi + 8 + i];
    }
    // B 32x16 f16, this code's contiguous k order: lanes 0-15 hold K = 0..15, lanes 16-31 hold K = 16..31
#pragma unroll
    for (int i = 0; i < 16; ++i) {
      bfrag[i] = WsT[ncol][k0 + 16 * hi + i];
    }
    acc = __builtin_amdgcn_wmma_f32_16x16x32_f16(
        /*neg_a=*/false, afrag, /*neg_b=*/false, bfrag,
        /*c_mod=*/(short)0, acc, /*reuse_a=*/false, /*reuse_b=*/false);
  }

  // ReLU + transpose through LDS for coalesced b128 stores.
#pragma unroll
  for (int v = 0; v < 8; ++v) {
    float r = acc[v];
    Outs[8 * hi + v][ncol] = (r > 0.0f) ? r : 0.0f;
  }
  __syncthreads();
  for (int i = tid; i < 16 * (DIM / 4); i += 256) {
    int r = i >> 5, c4 = (i & 31) * 4;
    const float4 o = make_float4(Outs[r][c4], Outs[r][c4 + 1],
                                 Outs[r][c4 + 2], Outs[r][c4 + 3]);
    *reinterpret_cast<float4*>(agg + (size_t)(row0 + r) * DIM + c4) = o;
  }
}

// ------------------------------------------------------------- mean pooling
__global__ __launch_bounds__(256) void pool_accum_kernel(const int* __restrict__ gid,
                                                         const float* __restrict__ h,
                                                         float* __restrict__ pooled,
                                                         float* __restrict__ counts,
                                                         int n_nodes) {
  int t = blockIdx.x * 256 + threadIdx.x;
  int i = t >> 5;
  if (i >= n_nodes) return;
  int q = (t & 31) * 4;
  int g = gid[i];
  if ((t & 31) == 0) atomicAdd(&counts[g], 1.0f);
  const float4 x = *reinterpret_cast<const float4*>(h + (size_t)i * DIM + q);
  float* out = pooled + (size_t)g * DIM + q;
  atomicAdd(out + 0, x.x);
  atomicAdd(out + 1, x.y);
  atomicAdd(out + 2, x.z);
  atomicAdd(out + 3, x.w);
}

// ----------------------------------------------------------- readout MLP
__global__ __launch_bounds__(256) void mlp_hidden_kernel(const float* __restrict__ pooled,
                                                         const float* __restrict__ counts,
                                                         const float* __restrict__ Wa,
                                                         const float* __restrict__ ba,
                                                         float* __restrict__ hidden) {
  int t = blockIdx.x * 256 + threadIdx.x;
  if (t >= N_GRAPHS * HID) return;
  int g = t / HID, j = t % HID;
  float inv = 1.0f / fmaxf(counts[g], 1.0f);
  float acc = ba[j];
  for (int k = 0; k < DIM; ++k) acc += (pooled[g * DIM + k] * inv) * Wa[k * HID + j];
  hidden[t] = fmaxf(acc, 0.0f);
}

__global__ __launch_bounds__(256) void mlp_out_kernel(const float* __restrict__ hidden,
                                                      const float* __restrict__ Wb,
                                                      const float* __restrict__ bb,
                                                      float* __restrict__ scores) {
  int t = blockIdx.x * 256 + threadIdx.x;
  if (t >= N_GRAPHS * N_CLASS) return;
  int g = t / N_CLASS, j = t % N_CLASS;
  float acc = bb[j];
  for (int k = 0; k < HID; ++k) acc += hidden[g * HID + k] * Wb[k * N_CLASS + j];
  scores[t] = acc;
}

// ------------------------------------------------------------------ launch
extern "C" void kernel_launch(void* const* d_in, const int* in_sizes, int n_in,
                              void* d_out, int out_size, void* d_ws, size_t ws_size,
                              hipStream_t stream) {
  const float* feat = (const float*)d_in[0];
  const int*   src  = (const int*)d_in[1];
  const int*   dst  = (const int*)d_in[2];
  const int*   gid  = (const int*)d_in[3];
  const float* Wl[6] = {(const float*)d_in[4],  (const float*)d_in[6],
                        (const float*)d_in[8],  (const float*)d_in[10],
                        (const float*)d_in[12], (const float*)d_in[14]};
  const float* bl[6] = {(const float*)d_in[5],  (const float*)d_in[7],
                        (const float*)d_in[9],  (const float*)d_in[11],
                        (const float*)d_in[13], (const float*)d_in[15]};
  const float* Wa = (const float*)d_in[16];
  const float* ba = (const float*)d_in[17];
  const float* Wb = (const float*)d_in[18];
  const float* bb = (const float*)d_in[19];

  float* ws     = (float*)d_ws;
  float* ns     = ws;                                   // [N]
  float* nd     = ws + N_NODES;                         // [N]
  float* buf1   = ws + 2 * (size_t)N_NODES;             // [N, DIM]
  float* pooled = buf1 + (size_t)N_NODES * DIM;         // [G, DIM]
  float* counts = pooled + N_GRAPHS * DIM;              // [G]
  float* hidden = counts + N_GRAPHS;                    // [G, HID]

  float* scores = (float*)d_out;                        // [G, N_CLASS] = 640
  float* bufA   = (float*)d_out + N_GRAPHS * N_CLASS;   // [N, DIM] h_agg region,
                                                        // reused as ping-pong buffer
  const int featElems = N_NODES * DIM;
  const int edgeWork  = N_EDGES * 32;   // one wave per edge
  const int nodeWork  = N_NODES * 32;

  // degrees -> normalization factors
  fill_zero4_kernel<<<(2 * N_NODES / 4 + 255) / 256, 256, 0, stream>>>(
      (float4*)ns, 2 * N_NODES / 4);
  degree_kernel<<<(N_EDGES + 255) / 256, 256, 0, stream>>>(src, dst, ns, nd, N_EDGES);
  norm_kernel<<<(N_NODES + 255) / 256, 256, 0, stream>>>(ns, nd, N_NODES);

  // 6 GraphConv layers, ping-ponging between bufA (inside d_out) and buf1
  const float* h = feat;
  float* bufs[2] = {bufA, buf1};
  for (int l = 0; l < 6; ++l) {
    float* agg = bufs[l & 1];
    fill_zero4_kernel<<<(featElems / 4 + 255) / 256, 256, 0, stream>>>(
        (float4*)agg, featElems / 4);
    edge_scatter_kernel<<<(edgeWork + 255) / 256, 256, 0, stream>>>(src, dst, ns, h, agg, N_EDGES);
    gemm_norm_bias_relu_kernel<<<N_NODES / 16, 256, 0, stream>>>(Wl[l], bl[l], nd, agg);
    h = agg;
  }
  // h == buf1 after layer 6

  // per-graph mean pooling + MLP readout
  fill_zero4_kernel<<<((N_GRAPHS * DIM + N_GRAPHS) / 4 + 255) / 256, 256, 0, stream>>>(
      (float4*)pooled, (N_GRAPHS * DIM + N_GRAPHS) / 4);
  pool_accum_kernel<<<(nodeWork + 255) / 256, 256, 0, stream>>>(gid, h, pooled, counts, N_NODES);
  mlp_hidden_kernel<<<(N_GRAPHS * HID + 255) / 256, 256, 0, stream>>>(pooled, counts, Wa, ba, hidden);
  mlp_out_kernel<<<(N_GRAPHS * N_CLASS + 255) / 256, 256, 0, stream>>>(hidden, Wb, bb, scores);

  // final unnormalized aggregation h_agg = segment_sum(h[src], dst)
  fill_zero4_kernel<<<(featElems / 4 + 255) / 256, 256, 0, stream>>>(
      (float4*)bufA, featElems / 4);
  edge_scatter_kernel<<<(edgeWork + 255) / 256, 256, 0, stream>>>(src, dst, nullptr, h, bufA, N_EDGES);
}
